// BugLocalizationGNN_50096498541274
// MI455X (gfx1250) — matrix operand order repaired and checked by reading, not compiled, not run on hardware
//
#include <hip/hip_runtime.h>
#include <hip/hip_bf16.h>

// ---------------------------------------------------------------------------
// Types for CDNA5 WMMA (wave32): V_WMMA_F32_16X16X32_BF16
// ---------------------------------------------------------------------------
typedef __attribute__((ext_vector_type(16))) __bf16 v16bf;
typedef __attribute__((ext_vector_type(8)))  float  v8f;

// Async global->LDS (CDNA5 GLOBAL_LOAD_ASYNC_TO_LDS_B128), guarded by probe
#ifdef __HIP_DEVICE_COMPILE__
#  if __has_builtin(__builtin_amdgcn_global_load_async_to_lds_b128)
#    define USE_ASYNC_LDS 1
#  endif
#endif
#ifndef USE_ASYNC_LDS
#  define USE_ASYNC_LDS 0
#endif

#if USE_ASYNC_LDS
// builtin parameter type per hipcc diagnostic: pointer to int4 (vector_size 16),
// global (AS1) source and LDS (AS3) destination
typedef int v4i_t __attribute__((vector_size(16)));
typedef __attribute__((address_space(1))) v4i_t* g_v4i_p;
typedef __attribute__((address_space(3))) v4i_t* l_v4i_p;
#endif

__device__ __forceinline__ unsigned short f32_to_bf16_rte(float f) {
    unsigned u = __float_as_uint(f);
    unsigned r = (u + 0x7FFFu + ((u >> 16) & 1u)) >> 16;
    return (unsigned short)r;
}

// Monotone float <-> uint encoding so atomicMax(u32) == float max
__device__ __forceinline__ unsigned enc_f32(float f) {
    unsigned u = __float_as_uint(f);
    return u ^ ((u >> 31) ? 0xFFFFFFFFu : 0x80000000u);
}
__device__ __forceinline__ float dec_f32(unsigned u) {
    unsigned b = (u & 0x80000000u) ? (u ^ 0x80000000u) : ~u;
    return __uint_as_float(b);
}

// ---------------------------------------------------------------------------
// fp32 -> bf16, 4 elements/thread
// ---------------------------------------------------------------------------
__global__ void k_f32_to_bf16_v4(const float* __restrict__ in,
                                 unsigned short* __restrict__ out, int n4) {
    int i = blockIdx.x * blockDim.x + threadIdx.x;
    if (i >= n4) return;
    float4 v = *(const float4*)(in + (size_t)i * 4);
    union { unsigned short u[4]; uint2 q; } t;
    t.u[0] = f32_to_bf16_rte(v.x);
    t.u[1] = f32_to_bf16_rte(v.y);
    t.u[2] = f32_to_bf16_rte(v.z);
    t.u[3] = f32_to_bf16_rte(v.w);
    *(uint2*)(out + (size_t)i * 4) = t.q;
}

// fp32 W[K x N] -> bf16 W^T[N x K] (weights are tiny; done once)
__global__ void k_w_to_bf16_t(const float* __restrict__ in,
                              unsigned short* __restrict__ out, int K, int N) {
    int i = blockIdx.x * blockDim.x + threadIdx.x;
    if (i >= K * N) return;
    int k = i / N, n = i % N;
    out[(size_t)n * K + k] = f32_to_bf16_rte(in[i]);
}

__global__ void k_fill_u32(unsigned* __restrict__ p, unsigned v, int n) {
    int i = blockIdx.x * blockDim.x + threadIdx.x;
    if (i < n) p[i] = v;
}

// ---------------------------------------------------------------------------
// bf16 WMMA GEMM: C[M x N] (f32) = A[M x K] x B (as B^T[N x K]), bf16 inputs.
// Block tile 64x128, 8 waves (2x4), wave tile 32x32, BK=64, double-buffered.
// Tile staging: async global->LDS (ASYNCcnt) when available, else reg pipeline.
// ---------------------------------------------------------------------------
#define BM 64
#define BN 128
#define BK 64
#define PAD 8

__global__ __launch_bounds__(256, 1) void k_gemm_bf16_wmma(
    const unsigned short* __restrict__ A,
    const unsigned short* __restrict__ Bt,
    float* __restrict__ C, int M, int K, int N)
{
    __shared__ unsigned short As[2][BM][BK + PAD];   // 144B row pitch (16B aligned)
    __shared__ unsigned short Bs[2][BN][BK + PAD];   // n-major (pre-transposed)

    const int tid  = threadIdx.x;
    const int wave = tid >> 5;
    const int lane = tid & 31;
    const int wm   = wave >> 2;      // 0..1
    const int wn   = wave & 3;       // 0..3
    const int lm   = lane & 15;
    const int lh   = lane >> 4;
    const int bm   = blockIdx.y * BM;
    const int bn   = blockIdx.x * BN;

    v8f acc[2][2] = {};
    const int nT = K / BK;

    // per-thread chunk coordinates (A: 2 x b128, B: 4 x b128)
    int arow[2], acol[2], brow[4], bcol[4];
    #pragma unroll
    for (int it = 0; it < 2; ++it) {
        int idx = tid + it * 256;
        arow[it] = idx >> 3; acol[it] = (idx & 7) * 8;
    }
    #pragma unroll
    for (int it = 0; it < 4; ++it) {
        int idx = tid + it * 256;
        brow[it] = idx >> 3; bcol[it] = (idx & 7) * 8;
    }

    auto compute = [&](int cur) {
        #pragma unroll
        for (int kk = 0; kk < BK; kk += 32) {
            union { uint4 q[2]; v16bf v; } af[2], bfg[2];
            #pragma unroll
            for (int i = 0; i < 2; ++i) {
                const unsigned short* ap = &As[cur][wm * 32 + i * 16 + lm][kk + lh * 8];
                af[i].q[0] = *(const uint4*)ap;           // K = kk+lh*8 .. +7
                af[i].q[1] = *(const uint4*)(ap + 16);    // K = kk+16+lh*8 .. +7
            }
            #pragma unroll
            for (int j = 0; j < 2; ++j) {
                const unsigned short* bp = &Bs[cur][wn * 32 + j * 16 + lm][kk + lh * 8];
                bfg[j].q[0] = *(const uint4*)bp;          // K = kk+lh*8 .. +7
                bfg[j].q[1] = *(const uint4*)(bp + 16);   // K = kk+16+lh*8 .. +7
            }
            #pragma unroll
            for (int i = 0; i < 2; ++i)
                #pragma unroll
                for (int j = 0; j < 2; ++j)
                    acc[i][j] = __builtin_amdgcn_wmma_f32_16x16x32_bf16(
                        false, af[i].v, false, bfg[j].v,
                        (short)0, acc[i][j], false, false);
        }
    };

#if USE_ASYNC_LDS
    // ---- async tensor-style pipeline: memory -> LDS, no VGPR staging ----
    auto issue = [&](int k0, int buf) {
        #pragma unroll
        for (int it = 0; it < 2; ++it) {
            int gr = bm + arow[it]; if (gr >= M) gr = M - 1;
            __builtin_amdgcn_global_load_async_to_lds_b128(
                (g_v4i_p)(A + (size_t)gr * K + k0 + acol[it]),
                (l_v4i_p)&As[buf][arow[it]][acol[it]], 0, 0);
        }
        #pragma unroll
        for (int it = 0; it < 4; ++it) {
            __builtin_amdgcn_global_load_async_to_lds_b128(
                (g_v4i_p)(Bt + (size_t)(bn + brow[it]) * K + k0 + bcol[it]),
                (l_v4i_p)&Bs[buf][brow[it]][bcol[it]], 0, 0);
        }
    };
    issue(0, 0);
    for (int t = 0; t < nT; ++t) {
        asm volatile("s_wait_asynccnt 0x0" ::: "memory");   // tile t landed in LDS
        __syncthreads();                                    // visible to all waves
        if (t + 1 < nT) issue((t + 1) * BK, (t + 1) & 1);   // overlaps WMMA below
        if (t + 2 < nT) {
            __builtin_prefetch(A  + (size_t)(bm + arow[0]) * K + (t + 2) * BK, 0, 1);
            __builtin_prefetch(Bt + (size_t)(bn + brow[0]) * K + (t + 2) * BK, 0, 1);
        }
        compute(t & 1);
    }
#else
    // ---- fallback: register-staged double buffer ----
    uint4 aR[2], bR[4];
    auto fetch = [&](int k0) {
        #pragma unroll
        for (int it = 0; it < 2; ++it) {
            int gr = bm + arow[it]; if (gr >= M) gr = M - 1;
            aR[it] = *(const uint4*)(A + (size_t)gr * K + k0 + acol[it]);
        }
        #pragma unroll
        for (int it = 0; it < 4; ++it)
            bR[it] = *(const uint4*)(Bt + (size_t)(bn + brow[it]) * K + k0 + bcol[it]);
    };
    auto stash = [&](int buf) {
        #pragma unroll
        for (int it = 0; it < 2; ++it) *(uint4*)&As[buf][arow[it]][acol[it]] = aR[it];
        #pragma unroll
        for (int it = 0; it < 4; ++it) *(uint4*)&Bs[buf][brow[it]][bcol[it]] = bR[it];
    };
    fetch(0); stash(0);
    __syncthreads();
    for (int t = 0; t < nT; ++t) {
        const bool pre = (t + 1 < nT);
        if (pre) fetch((t + 1) * BK);
        if (t + 2 < nT) {
            __builtin_prefetch(A  + (size_t)(bm + arow[0]) * K + (t + 2) * BK, 0, 1);
            __builtin_prefetch(Bt + (size_t)(bn + brow[0]) * K + (t + 2) * BK, 0, 1);
        }
        compute(t & 1);
        if (pre) stash((t + 1) & 1);
        __syncthreads();
    }
#endif

    // epilogue: fast path for full tiles, guarded only for the last partial one
    const bool full = (bm + BM <= M);
    #pragma unroll
    for (int i = 0; i < 2; ++i) {
        #pragma unroll
        for (int j = 0; j < 2; ++j) {
            int n  = bn + wn * 32 + j * 16 + lm;
            int m0 = bm + wm * 32 + i * 16 + lh * 8;
            float* p = C + (size_t)m0 * N + n;
            if (full) {
                #pragma unroll
                for (int v = 0; v < 8; ++v) { *p = acc[i][j][v]; p += N; }
            } else {
                #pragma unroll
                for (int v = 0; v < 8; ++v) {
                    if (m0 + v < M) *p = acc[i][j][v];
                    p += N;
                }
            }
        }
    }
}

// ---------------------------------------------------------------------------
// Per-node attention scores: one wave32 per (node, head); C == 128
// ---------------------------------------------------------------------------
__global__ __launch_bounds__(256) void k_node_scores(
    const float* __restrict__ h, const float* __restrict__ a_s,
    const float* __restrict__ a_d, float* __restrict__ s,
    float* __restrict__ d, int Nn, int H, int Cc)
{
    int warp = (blockIdx.x * blockDim.x + threadIdx.x) >> 5;
    int lane = threadIdx.x & 31;
    if (warp >= Nn * H) return;
    int node = warp / H, head = warp % H;
    float4 hv = *(const float4*)(h + (size_t)node * H * Cc + head * Cc + lane * 4);
    float4 as = *(const float4*)(a_s + head * Cc + lane * 4);
    float4 ad = *(const float4*)(a_d + head * Cc + lane * 4);
    float ss = hv.x * as.x + hv.y * as.y + hv.z * as.z + hv.w * as.w;
    float dd = hv.x * ad.x + hv.y * ad.y + hv.z * ad.z + hv.w * ad.w;
    #pragma unroll
    for (int off = 16; off >= 1; off >>= 1) {
        ss += __shfl_xor(ss, off, 32);
        dd += __shfl_xor(dd, off, 32);
    }
    if (lane == 0) { s[warp] = ss; d[warp] = dd; }
}

// ---------------------------------------------------------------------------
// Edge softmax phase 1: e = leaky_relu(s[src]+d[dst]); segment max via atomicMax
// ---------------------------------------------------------------------------
__global__ void k_edge_max(const int* __restrict__ ei, int E, int Nn, int H,
                           const float* __restrict__ s, const float* __restrict__ d,
                           float* __restrict__ ebuf, unsigned* __restrict__ menc)
{
    int gid = blockIdx.x * blockDim.x + threadIdx.x;
    int tot = (E + Nn) * H;
    if (gid >= tot) return;
    int e = gid / H, hh = gid % H;
    int sn, dn;
    if (e < E) { sn = ei[e]; dn = ei[E + e]; } else { sn = dn = e - E; }
    float v = s[sn * H + hh] + d[dn * H + hh];
    v = (v > 0.0f) ? v : 0.2f * v;                  // leaky_relu 0.2
    ebuf[gid] = v;
    atomicMax(&menc[dn * H + hh], enc_f32(v));
}

// phase 2: ex = exp(e - m[dst]); denom via atomicAdd
__global__ void k_edge_expsum(const int* __restrict__ ei, int E, int Nn, int H,
                              float* __restrict__ ebuf,
                              const unsigned* __restrict__ menc,
                              float* __restrict__ denom)
{
    int gid = blockIdx.x * blockDim.x + threadIdx.x;
    int tot = (E + Nn) * H;
    if (gid >= tot) return;
    int e = gid / H, hh = gid % H;
    int dn = (e < E) ? ei[E + e] : (e - E);
    float m  = dec_f32(menc[dn * H + hh]);
    float ex = __expf(ebuf[gid] - m);
    ebuf[gid] = ex;
    atomicAdd(&denom[dn * H + hh], ex);
}

// phase 3: agg[dst] += (ex/denom[dst]) * h[src]; thread per (edge, 4 feats)
__global__ __launch_bounds__(256) void k_edge_aggregate(
    const int* __restrict__ ei, int E, int Nn, int H, int Cc,
    const float* __restrict__ ebuf, const float* __restrict__ denom,
    const float* __restrict__ h, float* __restrict__ agg)
{
    int HC  = H * Cc;
    int tpe = HC >> 2;
    long long gid = (long long)blockIdx.x * blockDim.x + threadIdx.x;
    long long tot = (long long)(E + Nn) * tpe;
    if (gid >= tot) return;
    int e = (int)(gid / tpe);
    int c = (int)(gid % tpe) * 4;
    int hh = c / Cc;
    int sn, dn;
    if (e < E) { sn = ei[e]; dn = ei[E + e]; } else { sn = dn = e - E; }
    float alpha = ebuf[(size_t)e * H + hh] / denom[(size_t)dn * H + hh];
    float4 hv = *(const float4*)(h + (size_t)sn * HC + c);
    float* out = agg + (size_t)dn * HC + c;
    atomicAdd(out + 0, alpha * hv.x);
    atomicAdd(out + 1, alpha * hv.y);
    atomicAdd(out + 2, alpha * hv.z);
    atomicAdd(out + 3, alpha * hv.w);
}

// ---------------------------------------------------------------------------
// Fused bias + BN + ELU epilogue (4 channels/thread); optional f32/bf16 out
// ---------------------------------------------------------------------------
__global__ void k_finalize_v4(const float* __restrict__ agg, const float* __restrict__ b,
                              const float* __restrict__ g, const float* __restrict__ be,
                              const float* __restrict__ rm, const float* __restrict__ rv,
                              float* __restrict__ out_f32, unsigned short* __restrict__ out_bf16,
                              int Nn, int HC)
{
    long long gid = (long long)blockIdx.x * blockDim.x + threadIdx.x;
    long long tot = (long long)Nn * (HC >> 2);
    if (gid >= tot) return;
    int ch = (int)(gid % (HC >> 2)) * 4;
    float4 v  = *(const float4*)(agg + gid * 4);
    float4 bb = *(const float4*)(b  + ch);
    float4 gg = *(const float4*)(g  + ch);
    float4 eb = *(const float4*)(be + ch);
    float4 mm = *(const float4*)(rm + ch);
    float4 vv = *(const float4*)(rv + ch);
    float r[4];
    r[0] = (v.x + bb.x - mm.x) * (gg.x * __frsqrt_rn(vv.x + 1e-5f)) + eb.x;
    r[1] = (v.y + bb.y - mm.y) * (gg.y * __frsqrt_rn(vv.y + 1e-5f)) + eb.y;
    r[2] = (v.z + bb.z - mm.z) * (gg.z * __frsqrt_rn(vv.z + 1e-5f)) + eb.z;
    r[3] = (v.w + bb.w - mm.w) * (gg.w * __frsqrt_rn(vv.w + 1e-5f)) + eb.w;
    #pragma unroll
    for (int k = 0; k < 4; ++k) r[k] = (r[k] > 0.0f) ? r[k] : (__expf(r[k]) - 1.0f);
    if (out_f32) *(float4*)(out_f32 + gid * 4) = make_float4(r[0], r[1], r[2], r[3]);
    if (out_bf16) {
        union { unsigned short u[4]; uint2 q; } t;
        #pragma unroll
        for (int k = 0; k < 4; ++k) t.u[k] = f32_to_bf16_rte(r[k]);
        *(uint2*)(out_bf16 + gid * 4) = t.q;
    }
}

// ---------------------------------------------------------------------------
// Classifier: out[n,2] = h[n,:] @ Wc[128,2] + bc
// ---------------------------------------------------------------------------
__global__ void k_classifier(const float* __restrict__ h, const float* __restrict__ Wc,
                             const float* __restrict__ bc, float* __restrict__ out,
                             int Nn, int Cc)
{
    int n = blockIdx.x * blockDim.x + threadIdx.x;
    if (n >= Nn) return;
    float a0 = bc[0], a1 = bc[1];
    const float* hp = h + (size_t)n * Cc;
    for (int c = 0; c < Cc; c += 4) {
        float4 v = *(const float4*)(hp + c);
        a0 += v.x * Wc[(c+0)*2] + v.y * Wc[(c+1)*2] + v.z * Wc[(c+2)*2] + v.w * Wc[(c+3)*2];
        a1 += v.x * Wc[(c+0)*2+1] + v.y * Wc[(c+1)*2+1] + v.z * Wc[(c+2)*2+1] + v.w * Wc[(c+3)*2+1];
    }
    out[n * 2 + 0] = a0;
    out[n * 2 + 1] = a1;
}

// ---------------------------------------------------------------------------
// Host orchestration
// ---------------------------------------------------------------------------
static inline int cdiv(long long a, long long b) { return (int)((a + b - 1) / b); }

extern "C" void kernel_launch(void* const* d_in, const int* in_sizes, int n_in,
                              void* d_out, int out_size, void* d_ws, size_t ws_size,
                              hipStream_t stream) {
    const float* x   = (const float*)d_in[0];
    const int*   ei  = (const int*)d_in[1];      // [2,E] int32
    const float* W1  = (const float*)d_in[2];
    const float* a1s = (const float*)d_in[3];
    const float* a1d = (const float*)d_in[4];
    const float* b1  = (const float*)d_in[5];
    const float* g1  = (const float*)d_in[6];
    const float* be1 = (const float*)d_in[7];
    const float* rm1 = (const float*)d_in[8];
    const float* rv1 = (const float*)d_in[9];
    const float* W2  = (const float*)d_in[10];
    const float* a2s = (const float*)d_in[11];
    const float* a2d = (const float*)d_in[12];
    const float* b2  = (const float*)d_in[13];
    const float* g2  = (const float*)d_in[14];
    const float* be2 = (const float*)d_in[15];
    const float* rm2 = (const float*)d_in[16];
    const float* rv2 = (const float*)d_in[17];
    const float* W3  = (const float*)d_in[18];
    const float* a3s = (const float*)d_in[19];
    const float* a3d = (const float*)d_in[20];
    const float* b3  = (const float*)d_in[21];
    const float* g3  = (const float*)d_in[22];
    const float* be3 = (const float*)d_in[23];
    const float* rm3 = (const float*)d_in[24];
    const float* rv3 = (const float*)d_in[25];
    const float* Wc  = (const float*)d_in[26];
    const float* bc  = (const float*)d_in[27];
    float* out = (float*)d_out;

    const int HC  = in_sizes[5];            // H*HID = 512
    const int IN  = in_sizes[2] / HC;       // 768
    const int Nn  = in_sizes[0] / IN;       // 50000
    const int E   = in_sizes[1] / 2;        // 400000
    const int HID = in_sizes[21];           // 128
    const int H   = HC / HID;               // 4

    char* ws = (char*)d_ws;
    size_t off = 0;
    auto carve = [&](size_t bytes) -> char* {
        off = (off + 255) & ~(size_t)255;
        char* p = ws + off;
        off += bytes;
        return p;
    };
    unsigned short* xb  = (unsigned short*)carve((size_t)Nn * IN  * 2);
    unsigned short* W1t = (unsigned short*)carve((size_t)IN * HC  * 2);  // bf16 W1^T
    unsigned short* W2t = (unsigned short*)carve((size_t)HC * HC  * 2);
    unsigned short* W3t = (unsigned short*)carve((size_t)HC * HID * 2);
    float*          hh  = (float*)carve((size_t)Nn * HC * 4);
    unsigned short* hb  = (unsigned short*)carve((size_t)Nn * HC * 2);
    float*          agg = (float*)carve((size_t)Nn * HC * 4);
    float*          sb  = (float*)carve((size_t)Nn * H * 4);
    float*          db  = (float*)carve((size_t)Nn * H * 4);
    unsigned*       men = (unsigned*)carve((size_t)Nn * H * 4);
    float*          den = (float*)carve((size_t)Nn * H * 4);
    float*          exb = (float*)carve((size_t)(E + Nn) * H * 4);
    float*          h3  = (float*)carve((size_t)Nn * HID * 4);
    (void)ws_size;

    const int TB = 256;

    k_f32_to_bf16_v4<<<cdiv((long long)Nn * IN / 4, TB), TB, 0, stream>>>(x, xb, Nn * IN / 4);
    k_w_to_bf16_t<<<cdiv((long long)IN * HC, TB), TB, 0, stream>>>(W1, W1t, IN, HC);
    k_w_to_bf16_t<<<cdiv((long long)HC * HC, TB), TB, 0, stream>>>(W2, W2t, HC, HC);
    k_w_to_bf16_t<<<cdiv((long long)HC * HID, TB), TB, 0, stream>>>(W3, W3t, HC, HID);

    struct Layer {
        const unsigned short *Ain, *Wt;
        int K, Hh, Cc;
        const float *as, *ad, *bias, *g, *be, *rm, *rv;
        float* out_f32; unsigned short* out_bf16;
    };
    Layer layers[3] = {
        { xb, W1t, IN, H, HID, a1s, a1d, b1, g1, be1, rm1, rv1, nullptr, hb },
        { hb, W2t, HC, H, HID, a2s, a2d, b2, g2, be2, rm2, rv2, nullptr, hb },
        { hb, W3t, HC, 1, HID, a3s, a3d, b3, g3, be3, rm3, rv3, h3, nullptr },
    };

    for (int l = 0; l < 3; ++l) {
        const Layer& L = layers[l];
        const int lHC = L.Hh * L.Cc;

        dim3 ggrid(lHC / BN, cdiv(Nn, BM));
        k_gemm_bf16_wmma<<<ggrid, TB, 0, stream>>>(L.Ain, L.Wt, hh, Nn, L.K, lHC);

        k_node_scores<<<cdiv((long long)Nn * L.Hh * 32, TB), TB, 0, stream>>>(
            hh, L.as, L.ad, sb, db, Nn, L.Hh, L.Cc);

        k_fill_u32<<<cdiv((long long)Nn * lHC, TB), TB, 0, stream>>>((unsigned*)agg, 0u, Nn * lHC);
        k_fill_u32<<<cdiv((long long)Nn * L.Hh, TB), TB, 0, stream>>>(men, 0u, Nn * L.Hh);
        k_fill_u32<<<cdiv((long long)Nn * L.Hh, TB), TB, 0, stream>>>((unsigned*)den, 0u, Nn * L.Hh);

        int etot = (E + Nn) * L.Hh;
        k_edge_max<<<cdiv(etot, TB), TB, 0, stream>>>(ei, E, Nn, L.Hh, sb, db, exb, men);
        k_edge_expsum<<<cdiv(etot, TB), TB, 0, stream>>>(ei, E, Nn, L.Hh, exb, men, den);
        long long atot = (long long)(E + Nn) * (lHC >> 2);
        k_edge_aggregate<<<cdiv(atot, TB), TB, 0, stream>>>(
            ei, E, Nn, L.Hh, L.Cc, exb, den, hh, agg);

        k_finalize_v4<<<cdiv((long long)Nn * (lHC >> 2), TB), TB, 0, stream>>>(
            agg, L.bias, L.g, L.be, L.rm, L.rv, L.out_f32, L.out_bf16, Nn, lHC);
    }

    k_classifier<<<cdiv(Nn, TB), TB, 0, stream>>>(h3, Wc, bc, out, Nn, HID);
    (void)n_in; (void)out_size;
}
